// MultiScaleRetention_50010599194888
// MI455X (gfx1250) — matrix operand order repaired and checked by reading, not compiled, not run on hardware
//
#include <hip/hip_runtime.h>
#include <hip/hip_bf16.h>

// ---------------------------------------------------------------------------
// MultiScaleRetention on MI455X (gfx1250, wave32, WMMA bf16 16x16x32)
//   B=2, T=2048, D_MODEL=1024, H=16, Dh=64
// Strategy: bf16 inputs + f32 accumulation everywhere (8x K-depth per WMMA vs
// the 16x16x4 f32 op). Retention scores never touch HBM (flash-style fusion
// with the per-head exponential-decay causal mask).
// ---------------------------------------------------------------------------

#define BATCH 2
#define TSEQ  2048
#define DM    1024
#define NH    16
#define DH    64
#define BH    (BATCH * NH)      // 32
#define MTOT  (BATCH * TSEQ)    // 4096 rows
#define GN_EPS 1e-5f

typedef __bf16 bf16;
typedef bf16  v16bf __attribute__((ext_vector_type(16)));
typedef bf16  v8bf  __attribute__((ext_vector_type(8)));
typedef float v8f   __attribute__((ext_vector_type(8)));

#define SHUF16(a, b) __builtin_shufflevector((a), (b), 0,1,2,3,4,5,6,7,8,9,10,11,12,13,14,15)

__device__ __forceinline__ bf16 f2bf(float f) {
  union { float f; unsigned u; } v; v.f = f;
  unsigned r = v.u + 0x7FFFu + ((v.u >> 16) & 1u);   // round-to-nearest-even
  union { unsigned short u; bf16 b; } o; o.u = (unsigned short)(r >> 16);
  return o.b;
}

__device__ __forceinline__ v8f vzero8() {
  v8f z;
#pragma unroll
  for (int i = 0; i < 8; ++i) z[i] = 0.0f;
  return z;
}

// ---------------------------------------------------------------------------
// fp32 -> bf16 conversion (grid-stride)
// ---------------------------------------------------------------------------
__global__ void f32_to_bf16_kernel(const float* __restrict__ in,
                                   bf16* __restrict__ out, int n) {
  for (int i = blockIdx.x * blockDim.x + threadIdx.x; i < n;
       i += gridDim.x * blockDim.x)
    out[i] = f2bf(in[i]);
}

// ---------------------------------------------------------------------------
// V transpose: vb[B*T, DM] -> vt[(b*NH+h)*DH + d][T]  (bf16)
// ---------------------------------------------------------------------------
__global__ void transpose_v_kernel(const bf16* __restrict__ vb,
                                   bf16* __restrict__ vt, int n) {
  for (int o = blockIdx.x * blockDim.x + threadIdx.x; o < n;
       o += gridDim.x * blockDim.x) {
    int tr   = o % TSEQ;
    int rest = o / TSEQ;
    int d    = rest % DH;
    int bh   = rest / DH;
    int h    = bh % NH;
    int b    = bh / NH;
    vt[o] = vb[(size_t)(b * TSEQ + tr) * DM + h * DH + d];
  }
}

// ---------------------------------------------------------------------------
// Tiled WMMA GEMM: C[M,N] = A[M,K] * B[N,K]^T   (A,B bf16 row-major, K-major)
// Block tile 64x128, 8 waves (2x4), each wave 32x32 = 2x2 WMMA accumulators.
// ---------------------------------------------------------------------------
template <typename OutT>
__global__ __launch_bounds__(256) void wmma_gemm_kernel(
    const bf16* __restrict__ A, const bf16* __restrict__ Bm,
    OutT* __restrict__ C, int M, int N, int K) {
  __shared__ alignas(16) bf16 As[64][32];
  __shared__ alignas(16) bf16 Bs[128][32];

  const int tid  = threadIdx.x;
  const int lane = tid & 31;
  const int wave = tid >> 5;
  const int lh   = lane >> 4;   // half-wave: selects K sub-range
  const int ll   = lane & 15;   // row/col within fragment

  const int m0 = blockIdx.x * 64;
  const int n0 = blockIdx.y * 128;
  const int wm = (wave >> 2) * 32;   // wave tile inside block
  const int wn = (wave & 3) * 32;

  v8f acc[2][2];
#pragma unroll
  for (int i = 0; i < 2; ++i)
#pragma unroll
    for (int j = 0; j < 2; ++j) acc[i][j] = vzero8();

  for (int kk = 0; kk < K; kk += 32) {
    // cooperative loads: A tile 64x32 (8 bf16/thread), B tile 128x32 (16/thread)
    {
      int row = tid >> 2, kc = (tid & 3) * 8;
      *(v8bf*)&As[row][kc] =
          *(const v8bf*)(A + (size_t)(m0 + row) * K + kk + kc);
    }
    {
      int row = tid >> 1, kc = (tid & 1) * 16;
      const bf16* src = Bm + (size_t)(n0 + row) * K + kk + kc;
      *(v8bf*)&Bs[row][kc]     = *(const v8bf*)(src);
      *(v8bf*)&Bs[row][kc + 8] = *(const v8bf*)(src + 8);
    }
    if (kk + 32 < K) {  // hint next K tile into cache (global_prefetch_b8)
      __builtin_prefetch(A + (size_t)(m0 + (tid >> 2)) * K + kk + 32, 0, 1);
      __builtin_prefetch(Bm + (size_t)(n0 + (tid >> 1)) * K + kk + 32, 0, 1);
    }
    __syncthreads();

    // A fragments (16x32): lane = row M, 16 contiguous K elems
    v16bf af[2];
#pragma unroll
    for (int sm = 0; sm < 2; ++sm) {
      int m = wm + sm * 16 + ll;
      v8bf x0 = *(const v8bf*)&As[m][lh * 16];
      v8bf x1 = *(const v8bf*)&As[m][lh * 16 + 8];
      af[sm] = SHUF16(x0, x1);
    }
    // B fragments (32x16): lane = col N, 16 contiguous K elems
    v16bf bfr[2];
#pragma unroll
    for (int sn = 0; sn < 2; ++sn) {
      int n = wn + sn * 16 + ll;
      v8bf x0 = *(const v8bf*)&Bs[n][lh * 16];
      v8bf x1 = *(const v8bf*)&Bs[n][lh * 16 + 8];
      bfr[sn] = SHUF16(x0, x1);
    }
#pragma unroll
    for (int sm = 0; sm < 2; ++sm)
#pragma unroll
      for (int sn = 0; sn < 2; ++sn)
        acc[sm][sn] = __builtin_amdgcn_wmma_f32_16x16x32_bf16(
            false, af[sm], false, bfr[sn], (short)0, acc[sm][sn], false, false);
    __syncthreads();
  }

  // C layout: VGPR r -> row (r + lh*8), lane ll -> col
#pragma unroll
  for (int sm = 0; sm < 2; ++sm)
#pragma unroll
    for (int sn = 0; sn < 2; ++sn)
#pragma unroll
      for (int r = 0; r < 8; ++r) {
        int m = m0 + wm + sm * 16 + r + lh * 8;
        int n = n0 + wn + sn * 16 + ll;
        float v = acc[sm][sn][r];
        if constexpr (sizeof(OutT) == 2)
          C[(size_t)m * N + n] = (OutT)f2bf(v);
        else
          C[(size_t)m * N + n] = (OutT)v;
      }
}

// ---------------------------------------------------------------------------
// Retention core: one wave per 16-row tile per (b,h). Flash-style causal loop
// over 32-column pairs; decay mask fused; scores stay in LDS/VGPRs.
// ---------------------------------------------------------------------------
__global__ __launch_bounds__(256) void retention_kernel(
    const bf16* __restrict__ qb, const bf16* __restrict__ kb,
    const bf16* __restrict__ vt, float* __restrict__ yf,
    float* __restrict__ stats) {
  __shared__ alignas(16) bf16 sA[8][16][32];   // per-wave score staging

  const int lane = threadIdx.x & 31;
  const int wave = threadIdx.x >> 5;
  const int lh   = lane >> 4;
  const int ll   = lane & 15;

  const int gw = blockIdx.x * 8 + wave;
  const int tilesPerBH = TSEQ / 16;            // 128
  const int it = gw % tilesPerBH;
  const int bh = gw / tilesPerBH;              // 0..31
  const int h  = bh % NH;
  const int b  = bh / NH;
  const int i0 = it * 16;

  const float gamma = 1.0f - exp2f(-5.0f - (float)h);
  const float lg    = log2f(gamma);

  // Q A-fragments, dh=64 split into two K=32 steps
  v16bf qf[2];
  {
    const bf16* qrow = qb + (size_t)(b * TSEQ + i0 + ll) * DM + h * DH;
#pragma unroll
    for (int ks = 0; ks < 2; ++ks) {
      v8bf a0 = *(const v8bf*)(qrow + ks * 32 + lh * 16);
      v8bf a1 = *(const v8bf*)(qrow + ks * 32 + lh * 16 + 8);
      qf[ks] = SHUF16(a0, a1);
    }
  }

  v8f accy[4];
#pragma unroll
  for (int t = 0; t < 4; ++t) accy[t] = vzero8();

  for (int jp = 0; jp <= it; jp += 2) {
    // two 16x16 score tiles -> decayed bf16 in LDS (A-layout staging)
#pragma unroll
    for (int s = 0; s < 2; ++s) {
      const int j0 = (jp + s) * 16;
      v8f sc = vzero8();
      const bool live = (j0 <= i0);
      if (live) {
        const bf16* krow = kb + (size_t)(b * TSEQ + j0 + ll) * DM + h * DH;
#pragma unroll
        for (int ks = 0; ks < 2; ++ks) {
          v8bf b0 = *(const v8bf*)(krow + ks * 32 + lh * 16);
          v8bf b1 = *(const v8bf*)(krow + ks * 32 + lh * 16 + 8);
          v16bf kf = SHUF16(b0, b1);
          sc = __builtin_amdgcn_wmma_f32_16x16x32_bf16(
              false, qf[ks], false, kf, (short)0, sc, false, false);
        }
      }
#pragma unroll
      for (int r = 0; r < 8; ++r) {
        int m    = r + lh * 8;
        int diff = (i0 + m) - (j0 + ll);
        float v = (live && diff >= 0)
                      ? sc[r] * exp2f((float)diff * lg) * 0.125f  // /sqrt(64)
                      : 0.0f;
        sA[wave][m][s * 16 + ll] = f2bf(v);
      }
    }

    // re-read scores as a 16x32 A fragment (per-wave LDS, no barrier needed)
    v8bf a0 = *(const v8bf*)&sA[wave][ll][lh * 16];
    v8bf a1 = *(const v8bf*)&sA[wave][ll][lh * 16 + 8];
    v16bf af = SHUF16(a0, a1);

    // Y(16x64) += S(16x32) * V(32x64): 4 WMMAs over dh tiles
#pragma unroll
    for (int t = 0; t < 4; ++t) {
      const bf16* vrow =
          vt + ((size_t)bh * DH + t * 16 + ll) * TSEQ + jp * 16 + lh * 16;
      v8bf b0 = *(const v8bf*)vrow;
      v8bf b1 = *(const v8bf*)(vrow + 8);
      v16bf vfr = SHUF16(b0, b1);
      accy[t] = __builtin_amdgcn_wmma_f32_16x16x32_bf16(
          false, af, false, vfr, (short)0, accy[t], false, false);
    }
  }

  // store Y (f32) + accumulate GroupNorm statistics
  float s1 = 0.0f, s2 = 0.0f;
#pragma unroll
  for (int t = 0; t < 4; ++t)
#pragma unroll
    for (int r = 0; r < 8; ++r) {
      int m = r + lh * 8;
      int d = t * 16 + ll;
      float v = accy[t][r];
      yf[(size_t)(b * TSEQ + i0 + m) * DM + h * DH + d] = v;
      s1 += v;
      s2 += v * v;
    }
#pragma unroll
  for (int off = 16; off > 0; off >>= 1) {
    s1 += __shfl_xor(s1, off, 32);
    s2 += __shfl_xor(s2, off, 32);
  }
  if (lane == 0) {
    atomicAdd(&stats[bh * 2 + 0], s1);
    atomicAdd(&stats[bh * 2 + 1], s2);
  }
}

// ---------------------------------------------------------------------------
// GroupNorm helpers
// ---------------------------------------------------------------------------
__global__ void zero_stats_kernel(float* __restrict__ stats) {
  if (threadIdx.x < 2 * BH) stats[threadIdx.x] = 0.0f;
}

__global__ void finalize_stats_kernel(const float* __restrict__ stats,
                                      float* __restrict__ mrs) {
  int bh = threadIdx.x;
  if (bh < BH) {
    const float cnt = (float)(TSEQ * DH);
    float mean = stats[bh * 2 + 0] / cnt;
    float var  = stats[bh * 2 + 1] / cnt - mean * mean;
    mrs[bh * 2 + 0] = mean;
    mrs[bh * 2 + 1] = rsqrtf(var + GN_EPS);
  }
}

__global__ void norm_affine_kernel(const float* __restrict__ yf,
                                   const float* __restrict__ mrs,
                                   const float* __restrict__ gw,
                                   const float* __restrict__ gb,
                                   bf16* __restrict__ ynb, int n) {
  for (int i = blockIdx.x * blockDim.x + threadIdx.x; i < n;
       i += gridDim.x * blockDim.x) {
    int c   = i % DM;
    int row = i / DM;
    int b   = row / TSEQ;
    int h   = c / DH;
    int bh  = b * NH + h;
    float mean = mrs[bh * 2 + 0];
    float rstd = mrs[bh * 2 + 1];
    float v = (yf[i] - mean) * rstd * gw[c] + gb[c];
    ynb[i] = f2bf(v);
  }
}

// ---------------------------------------------------------------------------
// Host-side orchestration
// ---------------------------------------------------------------------------
extern "C" void kernel_launch(void* const* d_in, const int* in_sizes, int n_in,
                              void* d_out, int out_size, void* d_ws,
                              size_t ws_size, hipStream_t stream) {
  const float* x   = (const float*)d_in[0];
  const float* W_q = (const float*)d_in[1];
  const float* W_k = (const float*)d_in[2];
  const float* W_v = (const float*)d_in[3];
  const float* W_o = (const float*)d_in[4];
  const float* gnw = (const float*)d_in[5];
  const float* gnb = (const float*)d_in[6];
  float* out = (float*)d_out;

  // workspace carve-up (bytes)
  char* ws = (char*)d_ws;
  const size_t SZ_X  = (size_t)MTOT * DM * sizeof(bf16);  // 8 MiB
  const size_t SZ_W  = (size_t)DM * DM * sizeof(bf16);    // 2 MiB
  const size_t SZ_YF = (size_t)MTOT * DM * sizeof(float); // 16 MiB
  size_t off = 0;
  bf16* xb  = (bf16*)(ws + off); off += SZ_X;
  bf16* wqb = (bf16*)(ws + off); off += SZ_W;
  bf16* wkb = (bf16*)(ws + off); off += SZ_W;
  bf16* wvb = (bf16*)(ws + off); off += SZ_W;
  bf16* wob = (bf16*)(ws + off); off += SZ_W;
  bf16* qb  = (bf16*)(ws + off); off += SZ_X;
  bf16* kb  = (bf16*)(ws + off); off += SZ_X;
  bf16* vb  = (bf16*)(ws + off); off += SZ_X;
  bf16* vt  = (bf16*)(ws + off); off += SZ_X;
  float* yf = (float*)(ws + off); off += SZ_YF;
  bf16* ynb = (bf16*)(ws + off); off += SZ_X;
  float* stats = (float*)(ws + off); off += 256;
  float* mrs   = (float*)(ws + off); off += 256;

  const int nAct = MTOT * DM;  // 4M elems
  const int nW   = DM * DM;    // 1M elems

  // 1) bf16 conversions
  f32_to_bf16_kernel<<<1024, 256, 0, stream>>>(x, xb, nAct);
  f32_to_bf16_kernel<<<512, 256, 0, stream>>>(W_q, wqb, nW);
  f32_to_bf16_kernel<<<512, 256, 0, stream>>>(W_k, wkb, nW);
  f32_to_bf16_kernel<<<512, 256, 0, stream>>>(W_v, wvb, nW);
  f32_to_bf16_kernel<<<512, 256, 0, stream>>>(W_o, wob, nW);

  // 2) Q/K/V projections (C = x * W^T), bf16 out
  dim3 ggrid(MTOT / 64, DM / 128);  // 64 x 8
  wmma_gemm_kernel<bf16><<<ggrid, 256, 0, stream>>>(xb, wqb, qb, MTOT, DM, DM);
  wmma_gemm_kernel<bf16><<<ggrid, 256, 0, stream>>>(xb, wkb, kb, MTOT, DM, DM);
  wmma_gemm_kernel<bf16><<<ggrid, 256, 0, stream>>>(xb, wvb, vb, MTOT, DM, DM);

  // 3) V transpose for contiguous S*V fragment loads
  transpose_v_kernel<<<2048, 256, 0, stream>>>(vb, vt, nAct);

  // 4) retention core + GroupNorm stats
  zero_stats_kernel<<<1, 64, 0, stream>>>(stats);
  const int totalWaves = BH * (TSEQ / 16);  // 4096
  retention_kernel<<<totalWaves / 8, 256, 0, stream>>>(qb, kb, vt, yf, stats);

  // 5) GroupNorm finalize + normalize/affine -> bf16
  finalize_stats_kernel<<<1, 32, 0, stream>>>(stats, mrs);
  norm_affine_kernel<<<2048, 256, 0, stream>>>(yf, mrs, gnw, gnb, ynb, nAct);

  // 6) output projection, fp32 -> d_out
  wmma_gemm_kernel<float><<<ggrid, 256, 0, stream>>>(ynb, wob, out, MTOT, DM, DM);
}
